// VRWKV_SpatialMix_12249246728671
// MI455X (gfx1250) — matrix operand order repaired and checked by reading, not compiled, not run on hardware
//
#include <hip/hip_runtime.h>
#include <hip/hip_bf16.h>

typedef __bf16 bf16;
typedef __attribute__((ext_vector_type(16))) __bf16        v16bf;
typedef __attribute__((ext_vector_type(8)))  float         v8f;
typedef __attribute__((ext_vector_type(4)))  unsigned int  u32x4;

constexpr int BB = 16, TT = 1024, CC = 768;
constexpr int HHc = 32, WWc = 32;
constexpr int MM = BB * TT;   // 16384 GEMM rows
constexpr int KK = CC;        // 768
constexpr int NN = CC;        // 768

// Async global->LDS path (CDNA5): direct copy into LDS, tracked by ASYNCcnt.
#if __has_builtin(__builtin_amdgcn_global_load_async_to_lds_b128) && \
    __has_builtin(__builtin_amdgcn_s_wait_asynccnt)
#define USE_ASYNC_LDS 1
#else
#define USE_ASYNC_LDS 0
#endif

#if USE_ASYNC_LDS
// Exact pointee type from the clang diagnostic: vector_size(16) int,
// in AS1 (printed as __device__) for global, AS3 for LDS.
typedef int vsi4 __attribute__((vector_size(16)));
typedef __attribute__((address_space(1))) vsi4 as1_v4i;
typedef __attribute__((address_space(3))) vsi4 as3_v4i;
#endif

// -------------------------------------------------------------------------
// Kernel 1: fused q_shift + token mix, f32 -> bf16 (A operands for 3 GEMMs)
// -------------------------------------------------------------------------
__global__ __launch_bounds__(256)
void mix_shift_kernel(const float* __restrict__ x,
                      const float* __restrict__ mk,
                      const float* __restrict__ mv,
                      const float* __restrict__ mr,
                      bf16* __restrict__ xk,
                      bf16* __restrict__ xv,
                      bf16* __restrict__ xr)
{
    size_t i = (size_t)blockIdx.x * blockDim.x + threadIdx.x;
    if (i >= (size_t)MM * CC) return;
    int c  = (int)(i % CC);
    size_t bt = i / CC;
    int t  = (int)(bt % TT);
    int h  = t >> 5;
    int w  = t & 31;
    int grp = c / (CC / 4);
    float xx = 0.0f;
    if (grp == 0)      { if (w >= 1)       xx = x[i - CC]; }               // from w-1
    else if (grp == 1) { if (w <= WWc - 2) xx = x[i + CC]; }               // from w+1
    else if (grp == 2) { if (h >= 1)       xx = x[i - (size_t)WWc * CC]; } // from h-1
    else               { if (h <= HHc - 2) xx = x[i + (size_t)WWc * CC]; } // from h+1
    float xs = x[i];
    float a = mk[c], b = mv[c], r = mr[c];
    xk[i] = (bf16)(xs * a + xx * (1.0f - a));
    xv[i] = (bf16)(xs * b + xx * (1.0f - b));
    xr[i] = (bf16)(xs * r + xx * (1.0f - r));
}

// -------------------------------------------------------------------------
// Kernel 2: weight convert f32 -> bf16 (weights then live in L2: 192 MB)
// -------------------------------------------------------------------------
__global__ __launch_bounds__(256)
void cvt_bf16_kernel(const float* __restrict__ in, bf16* __restrict__ out, int n)
{
    int i = blockIdx.x * blockDim.x + threadIdx.x;
    if (i < n) out[i] = (bf16)in[i];
}

// -------------------------------------------------------------------------
// Kernel 3: WMMA bf16 GEMM:  Out[M,N] = A[M,K] * W[N,K]^T  (f32 accumulate)
// 256 threads = 8 waves; block tile 128(M) x 64(N); K step 32.
// Wave grid 4(M) x 2(N): each wave owns a 32x32 patch = 2x2 WMMA tiles.
// Staging: GLOBAL_LOAD_ASYNC_TO_LDS_B128 (ASYNCcnt) when available.
// mode 0: store raw f32; mode 1: store sigmoid(f32)
// -------------------------------------------------------------------------
constexpr int TM  = 128;
constexpr int TN  = 64;
constexpr int TK  = 32;
constexpr int LDA = 48;   // padded LDS row stride (bf16 elems), 96B = 16B aligned
constexpr int LDB = 48;

union Frag { v16bf v; u32x4 q[2]; };

__global__ __launch_bounds__(256)
void wmma_gemm_kernel(const bf16* __restrict__ A,
                      const bf16* __restrict__ W,
                      float* __restrict__ Out,
                      int mode)
{
    __shared__ __align__(16) bf16 sA[TM * LDA];
    __shared__ __align__(16) bf16 sB[TN * LDB];

    const int tid  = threadIdx.x;
    const int n0   = blockIdx.x * TN;
    const int m0   = blockIdx.y * TM;
    const int wave = tid >> 5;
    const int lane = tid & 31;
    const int wm   = wave & 3;   // 4 waves along M
    const int wn   = wave >> 2;  // 2 waves along N
    const int half = lane >> 4;  // which 16-lane half
    const int l16  = lane & 15;

    // staging geometry: 16B chunks; A = two chunks/thread, B = one chunk/thread
    const int arow = tid >> 2;          // 0..63
    const int acb  = (tid & 3) * 8;     // bf16 column offset of 8-elem chunk

    v8f acc[2][2] = {};

    for (int k0 = 0; k0 < KK; k0 += TK) {
        // prefetch next K tile into cache (emits global_prefetch_b8)
        if (k0 + TK < KK) {
            __builtin_prefetch(&A[(size_t)(m0 + arow) * KK + k0 + TK + acb], 0, 0);
            if (tid < TN * 4)
                __builtin_prefetch(&W[(size_t)(n0 + arow) * KK + k0 + TK + acb], 0, 0);
        }
        __syncthreads();  // prior iteration's ds reads complete before overwrite

#if USE_ASYNC_LDS
        // A tile: 128 rows x 32 bf16 = 512 x 16B chunks, 2 per thread
        __builtin_amdgcn_global_load_async_to_lds_b128(
            (as1_v4i*)&A[(size_t)(m0 + arow) * KK + k0 + acb],
            (as3_v4i*)&sA[arow * LDA + acb], 0, 0);
        __builtin_amdgcn_global_load_async_to_lds_b128(
            (as1_v4i*)&A[(size_t)(m0 + arow + 64) * KK + k0 + acb],
            (as3_v4i*)&sA[(arow + 64) * LDA + acb], 0, 0);
        // B tile: 64 rows x 32 bf16 = 256 x 16B chunks, 1 per thread
        __builtin_amdgcn_global_load_async_to_lds_b128(
            (as1_v4i*)&W[(size_t)(n0 + arow) * KK + k0 + acb],
            (as3_v4i*)&sB[arow * LDB + acb], 0, 0);
        __builtin_amdgcn_s_wait_asynccnt(0);
#else
        #pragma unroll
        for (int rep = 0; rep < 2; ++rep) {
            int row = arow + rep * 64;
            *(u32x4*)&sA[row * LDA + acb] =
                *(const u32x4*)&A[(size_t)(m0 + row) * KK + k0 + acb];
        }
        *(u32x4*)&sB[arow * LDB + acb] =
            *(const u32x4*)&W[(size_t)(n0 + arow) * KK + k0 + acb];
#endif
        __syncthreads();

        // A fragments: lane holds row (l16), K runs [half*8, half*8+8) and +16
        Frag a[2], b[2];
        #pragma unroll
        for (int sm = 0; sm < 2; ++sm) {
            int row = wm * 32 + sm * 16 + l16;
            a[sm].q[0] = *(const u32x4*)&sA[row * LDA + half * 8];
            a[sm].q[1] = *(const u32x4*)&sA[row * LDA + 16 + half * 8];
        }
        // B fragments: lane holds weight row (col of B) l16, K runs [half*8, half*8+8) and +16
        #pragma unroll
        for (int sn = 0; sn < 2; ++sn) {
            int row = wn * 32 + sn * 16 + l16;
            b[sn].q[0] = *(const u32x4*)&sB[row * LDB + half * 8];
            b[sn].q[1] = *(const u32x4*)&sB[row * LDB + 16 + half * 8];
        }
        #pragma unroll
        for (int sm = 0; sm < 2; ++sm)
            #pragma unroll
            for (int sn = 0; sn < 2; ++sn)
                acc[sm][sn] = __builtin_amdgcn_wmma_f32_16x16x32_bf16(
                    false, a[sm].v, false, b[sn].v,
                    (short)0, acc[sm][sn], false, false);
    }

    // epilog: D layout — lane l16 = N, vgpr r -> M = r + half*8
    #pragma unroll
    for (int sm = 0; sm < 2; ++sm) {
        #pragma unroll
        for (int sn = 0; sn < 2; ++sn) {
            #pragma unroll
            for (int r = 0; r < 8; ++r) {
                int m = m0 + wm * 32 + sm * 16 + half * 8 + r;
                int n = n0 + wn * 32 + sn * 16 + l16;
                float val = acc[sm][sn][r];
                if (mode == 1) val = 1.0f / (1.0f + __expf(-val));
                Out[(size_t)m * NN + n] = val;
            }
        }
    }
}

// -------------------------------------------------------------------------
// Kernel 4: backward exclusive scan of bi_wkv + per-(b,c) kmax
// one thread per (b,c); consecutive c -> coalesced accesses each t step
// -------------------------------------------------------------------------
__global__ __launch_bounds__(256)
void bwd_scan_kernel(const float* __restrict__ k,
                     const float* __restrict__ v,
                     const float* __restrict__ sdecay,
                     float* __restrict__ bn,
                     float* __restrict__ bd,
                     float* __restrict__ kmaxbuf)
{
    int idx = blockIdx.x * blockDim.x + threadIdx.x;
    if (idx >= BB * CC) return;
    int c = idx % CC;
    int b = idx / CC;
    const float ew   = __expf(-__expf(sdecay[c] * (1.0f / TT)));
    const size_t base = (size_t)b * TT * CC + c;

    float km = -3.4e38f;
    #pragma unroll 4
    for (int t = 0; t < TT; ++t) km = fmaxf(km, k[base + (size_t)t * CC]);
    kmaxbuf[idx] = km;

    float num = 0.0f, den = 0.0f;
    for (int t = TT - 1; t >= 0; --t) {
        size_t o = base + (size_t)t * CC;
        bn[o] = num;
        bd[o] = den;
        float ek  = __expf(k[o] - km);
        float ekv = ek * v[o];
        num = ew * num + ekv;
        den = ew * den + ek;
    }
}

// -------------------------------------------------------------------------
// Kernel 5: forward scan + combine + sr-gate, emits bf16 A-operand for the
// output projection GEMM:  a = bf16( sigmoid(r) * rwkv )
// -------------------------------------------------------------------------
__global__ __launch_bounds__(256)
void fwd_combine_kernel(const float* __restrict__ k,
                        const float* __restrict__ v,
                        const float* __restrict__ sr,
                        const float* __restrict__ bn,
                        const float* __restrict__ bd,
                        const float* __restrict__ kmaxbuf,
                        const float* __restrict__ sdecay,
                        const float* __restrict__ sfirst,
                        bf16* __restrict__ aout)
{
    int idx = blockIdx.x * blockDim.x + threadIdx.x;
    if (idx >= BB * CC) return;
    int c = idx % CC;
    int b = idx / CC;
    const float ew = __expf(-__expf(sdecay[c] * (1.0f / TT)));
    const float u  = sfirst[c] * (1.0f / TT);
    const float km = kmaxbuf[idx];
    const size_t base = (size_t)b * TT * CC + c;

    float fn = 0.0f, fd = 0.0f;
    for (int t = 0; t < TT; ++t) {
        size_t o = base + (size_t)t * CC;
        float kk = k[o];
        float vv = v[o];
        float eu = __expf(u + kk - km);
        float rw = (fn + bn[o] + eu * vv) / (fd + bd[o] + eu);
        aout[o]  = (bf16)(sr[o] * rw);
        float ek  = __expf(kk - km);
        fn = ew * fn + ek * vv;
        fd = ew * fd + ek;
    }
}

// -------------------------------------------------------------------------
// Launch
// -------------------------------------------------------------------------
extern "C" void kernel_launch(void* const* d_in, const int* in_sizes, int n_in,
                              void* d_out, int out_size, void* d_ws, size_t ws_size,
                              hipStream_t stream)
{
    (void)in_sizes; (void)n_in; (void)out_size; (void)ws_size;

    const float* x      = (const float*)d_in[0];
    const float* key_w  = (const float*)d_in[1];
    const float* val_w  = (const float*)d_in[2];
    const float* rec_w  = (const float*)d_in[3];
    const float* out_w  = (const float*)d_in[4];
    const float* sdecay = (const float*)d_in[5];
    const float* sfirst = (const float*)d_in[6];
    const float* mix_k  = (const float*)d_in[7];
    const float* mix_v  = (const float*)d_in[8];
    const float* mix_r  = (const float*)d_in[9];
    float* out = (float*)d_out;

    // ---- workspace carve (all regions written before read) ----
    const size_t MTC   = (size_t)MM * CC;       // 12,582,912 elems
    const size_t SZ_BF = MTC * sizeof(bf16);    // 25.2 MB
    const size_t SZ_F  = MTC * sizeof(float);   // 50.3 MB
    const size_t SZ_W  = (size_t)CC * CC * sizeof(bf16);
    char* p = (char*)d_ws;
    bf16*  xk    = (bf16*)p;  p += SZ_BF;
    bf16*  xv    = (bf16*)p;  p += SZ_BF;
    bf16*  xr    = (bf16*)p;  p += SZ_BF;
    bf16*  wk    = (bf16*)p;  p += SZ_W;
    bf16*  wv    = (bf16*)p;  p += SZ_W;
    bf16*  wr    = (bf16*)p;  p += SZ_W;
    bf16*  wo    = (bf16*)p;  p += SZ_W;
    float* kbuf  = (float*)p; p += SZ_F;
    float* vbuf  = (float*)p; p += SZ_F;
    float* srbuf = (float*)p; p += SZ_F;
    float* bnbuf = (float*)p; p += SZ_F;
    float* bdbuf = (float*)p; p += SZ_F;
    float* kmaxb = (float*)p; p += (size_t)BB * CC * sizeof(float);
    bf16*  abuf  = (bf16*)p;  p += SZ_BF;

    // 1) q_shift + mixes -> bf16
    {
        int blocks = (int)((MTC + 255) / 256);
        mix_shift_kernel<<<blocks, 256, 0, stream>>>(x, mix_k, mix_v, mix_r, xk, xv, xr);
    }
    // 2) weights f32 -> bf16
    {
        int n = CC * CC;
        int blocks = (n + 255) / 256;
        cvt_bf16_kernel<<<blocks, 256, 0, stream>>>(key_w, wk, n);
        cvt_bf16_kernel<<<blocks, 256, 0, stream>>>(val_w, wv, n);
        cvt_bf16_kernel<<<blocks, 256, 0, stream>>>(rec_w, wr, n);
        cvt_bf16_kernel<<<blocks, 256, 0, stream>>>(out_w, wo, n);
    }
    // 3) three WMMA GEMMs: k, v, sigmoid(r)
    {
        dim3 grid(NN / TN, MM / TM);   // (12, 128)
        wmma_gemm_kernel<<<grid, 256, 0, stream>>>(xk, wk, kbuf,  0);
        wmma_gemm_kernel<<<grid, 256, 0, stream>>>(xv, wv, vbuf,  0);
        wmma_gemm_kernel<<<grid, 256, 0, stream>>>(xr, wr, srbuf, 1);
    }
    // 4) bidirectional WKV scan
    {
        int blocks = (BB * CC + 255) / 256;   // 48
        bwd_scan_kernel<<<blocks, 256, 0, stream>>>(kbuf, vbuf, sdecay, bnbuf, bdbuf, kmaxb);
        fwd_combine_kernel<<<blocks, 256, 0, stream>>>(kbuf, vbuf, srbuf, bnbuf, bdbuf,
                                                       kmaxb, sdecay, sfirst, abuf);
    }
    // 5) output projection GEMM -> d_out (f32)
    {
        dim3 grid(NN / TN, MM / TM);
        wmma_gemm_kernel<<<grid, 256, 0, stream>>>(abuf, wo, out, 0);
    }
}
